// BlockwiseParallelTransformerAttention_39075612459198
// MI455X (gfx1250) — matrix operand order repaired and checked by reading, not compiled, not run on hardware
//
#include <hip/hip_runtime.h>
#include <hip/hip_bf16.h>

// ---------------------------------------------------------------------------
// Problem constants (reference: B=2, S=2048, D=1024, H=16, Dh=64)
// ---------------------------------------------------------------------------
#define BATCH 2
#define SLEN  2048
#define DDIM  1024
#define HNUM  16
#define DHEAD 64
#define MTOT  (BATCH * SLEN)   // 4096 rows for the big GEMMs

typedef __attribute__((ext_vector_type(16))) __bf16          bf16x16;
typedef __attribute__((ext_vector_type(8)))  float           f32x8;
typedef __attribute__((ext_vector_type(8)))  unsigned short  u16x8;
typedef __attribute__((ext_vector_type(16))) unsigned short  u16x16;

// ---------------------------------------------------------------------------
// fp32 -> bf16 (round to nearest even), stored as raw u16 bits
// ---------------------------------------------------------------------------
__device__ __forceinline__ unsigned short f32_to_bf16(float f) {
  union { float f; unsigned int u; } v; v.f = f;
  unsigned int u = v.u;
  u += 0x7FFFu + ((u >> 16) & 1u);
  return (unsigned short)(u >> 16);
}

// ---------------------------------------------------------------------------
// WMMA bf16 fragment loaders (CDNA5 wave32 layouts, CDNA5 ISA §7.12.2)
//
// A fragment (16x32, MxK): lane holds row (lane&15); element j holds
//   k = 8*(lane>>4) + (j<8 ? j : 8+j)   -> two contiguous 16B loads (+0, +16)
// B fragment (32x16, KxN) with B[k][n] = src[row0+n][k0+k] (src row-major,
//   k contiguous): lane holds column n=(lane&15); element j holds
//   k = 8*(lane>>4) + (j<8 ? j : 8+j)   -> two contiguous 16B loads (+0, +16)
// ---------------------------------------------------------------------------
__device__ __forceinline__ u16x16 load_a_frag(const unsigned short* base,
                                              int stride, int row0, int k0,
                                              int lane) {
  const unsigned short* p =
      base + (size_t)(row0 + (lane & 15)) * (size_t)stride + k0 + ((lane >> 4) << 3);
  u16x8 lo = *(const u16x8*)(p);
  u16x8 hi = *(const u16x8*)(p + 16);
  return __builtin_shufflevector(lo, hi, 0, 1, 2, 3, 4, 5, 6, 7,
                                         8, 9, 10, 11, 12, 13, 14, 15);
}

__device__ __forceinline__ u16x16 load_b_frag(const unsigned short* base,
                                              int stride, int row0, int k0,
                                              int lane) {
  const unsigned short* p =
      base + (size_t)(row0 + (lane & 15)) * (size_t)stride + k0 + ((lane >> 4) << 3);
  u16x8 lo = *(const u16x8*)(p);
  u16x8 hi = *(const u16x8*)(p + 16);
  return __builtin_shufflevector(lo, hi, 0, 1, 2, 3, 4, 5, 6, 7,
                                         8, 9, 10, 11, 12, 13, 14, 15);
}

__device__ __forceinline__ f32x8 wmma_bf16(u16x16 a, u16x16 b, f32x8 c) {
  return __builtin_amdgcn_wmma_f32_16x16x32_bf16(
      /*neg_a=*/false, __builtin_bit_cast(bf16x16, a),
      /*neg_b=*/false, __builtin_bit_cast(bf16x16, b),
      /*c_mod=*/(short)0, c, /*reuse_a=*/false, /*reuse_b=*/false);
}

// ---------------------------------------------------------------------------
// Kernel 1: fp32 -> bf16 conversion (grid-stride)
// ---------------------------------------------------------------------------
__global__ void cvt_bf16_kernel(const float* __restrict__ in,
                                unsigned short* __restrict__ out, int n) {
  int i = blockIdx.x * blockDim.x + threadIdx.x;
  int stride = gridDim.x * blockDim.x;
  for (; i < n; i += stride) out[i] = f32_to_bf16(in[i]);
}

// ---------------------------------------------------------------------------
// Kernel 2: C = A(MxK) @ W(NxK)^T  via bf16 WMMA, f32 accumulate.
// Block tile 128x128, 8 waves (256 thr): waves 4(M) x 2(N), each wave a
// 32x64 tile (2x4 WMMA frags) -> 8 WMMA per 6 fragment loads per k-step.
//   mode 0: out bf16 row-major (Q with scale=1/sqrt(D), K with scale=1)
//   mode 1: out bf16 transposed per head -> Vt[B][H][Dh][S]
//   mode 2: out fp32 row-major + bias    (final feed-forward)
// ---------------------------------------------------------------------------
__global__ __launch_bounds__(256) void gemm_bf16_kernel(
    const unsigned short* __restrict__ A, const unsigned short* __restrict__ W,
    const float* __restrict__ bias, unsigned short* __restrict__ outb,
    float* __restrict__ outf, int M, int N, int K, int mode, float scale) {
  const int lane = threadIdx.x & 31;
  const int wv   = threadIdx.x >> 5;
  const int wm   = wv & 3;            // waves: 4 along M x 2 along N
  const int wn   = wv >> 2;
  const int m0   = blockIdx.x * 128 + wm * 32;
  const int n0   = blockIdx.y * 128 + wn * 64;

  f32x8 acc[2][4] = {};
#pragma unroll 4
  for (int kt = 0; kt < K; kt += 32) {
    u16x16 a0 = load_a_frag(A, K, m0,      kt, lane);
    u16x16 a1 = load_a_frag(A, K, m0 + 16, kt, lane);
    u16x16 b0 = load_b_frag(W, K, n0,      kt, lane);
    u16x16 b1 = load_b_frag(W, K, n0 + 16, kt, lane);
    u16x16 b2 = load_b_frag(W, K, n0 + 32, kt, lane);
    u16x16 b3 = load_b_frag(W, K, n0 + 48, kt, lane);
    acc[0][0] = wmma_bf16(a0, b0, acc[0][0]);
    acc[0][1] = wmma_bf16(a0, b1, acc[0][1]);
    acc[0][2] = wmma_bf16(a0, b2, acc[0][2]);
    acc[0][3] = wmma_bf16(a0, b3, acc[0][3]);
    acc[1][0] = wmma_bf16(a1, b0, acc[1][0]);
    acc[1][1] = wmma_bf16(a1, b1, acc[1][1]);
    acc[1][2] = wmma_bf16(a1, b2, acc[1][2]);
    acc[1][3] = wmma_bf16(a1, b3, acc[1][3]);
  }

  // C/D layout: lane holds column n=(lane&15); element e holds row 8*(lane>>4)+e
  const int cb     = lane & 15;
  const int rowoff = (lane >> 4) * 8;
#pragma unroll
  for (int i = 0; i < 2; ++i) {
#pragma unroll
    for (int j = 0; j < 4; ++j) {
      const int col = n0 + j * 16 + cb;
      const float bv = (mode == 2) ? bias[col] : 0.0f;
#pragma unroll
      for (int e = 0; e < 8; ++e) {
        const int row = m0 + i * 16 + rowoff + e;
        const float v = acc[i][j][e] * scale;
        if (mode == 0) {
          outb[(size_t)row * N + col] = f32_to_bf16(v);
        } else if (mode == 1) {
          const int b = row >> 11, s = row & (SLEN - 1);
          const int h = col >> 6,  d = col & (DHEAD - 1);
          outb[(((size_t)(b * HNUM + h) * DHEAD + d) << 11) + s] = f32_to_bf16(v);
        } else {
          outf[(size_t)row * N + col] = v + bv;
        }
      }
    }
  }
}

// ---------------------------------------------------------------------------
// Kernel 3: flash attention, one wave owns 16 query rows; streams keys in
// chunks of 32 with online softmax. 8 waves/block -> 128 query rows/block.
// grid = (S/128, H, B).
// ---------------------------------------------------------------------------
__global__ __launch_bounds__(256) void flash_attn_kernel(
    const unsigned short* __restrict__ Q, const unsigned short* __restrict__ Km,
    const unsigned short* __restrict__ Vt, unsigned short* __restrict__ O) {
  __shared__ unsigned short lds_p[8][16][32];   // per-wave P tile (bf16)

  const int lane = threadIdx.x & 31;
  const int wv   = threadIdx.x >> 5;
  const int b    = blockIdx.z;
  const int h    = blockIdx.y;
  const int q0   = blockIdx.x * 128 + wv * 16;

  const unsigned short* Qb = Q  + (size_t)b * SLEN * DDIM + h * DHEAD;
  const unsigned short* Kb = Km + (size_t)b * SLEN * DDIM + h * DHEAD;
  const unsigned short* Vb = Vt + (size_t)(b * HNUM + h) * DHEAD * SLEN;

  // Q fragments for this wave's 16 rows (Dh=64 -> 2 K-chunks of 32), kept live
  u16x16 qf[2];
  qf[0] = load_a_frag(Qb, DDIM, q0, 0,  lane);
  qf[1] = load_a_frag(Qb, DDIM, q0, 32, lane);

  f32x8 of[4] = {};      // out accumulators, 16 rows x 64 d
  f32x8 mrun, lsum;
#pragma unroll
  for (int e = 0; e < 8; ++e) { mrun[e] = -3.0e38f; lsum[e] = 0.0f; }

#pragma unroll 1
  for (int key0 = 0; key0 < SLEN; key0 += 32) {
    // ---- scores: 16x32 tile = two 16x16 WMMA n-tiles, Dh reduced in 2 steps
    f32x8 s0 = {}, s1 = {};
#pragma unroll
    for (int dc = 0; dc < 2; ++dc) {
      u16x16 k0f = load_b_frag(Kb, DDIM, key0,      dc * 32, lane);
      u16x16 k1f = load_b_frag(Kb, DDIM, key0 + 16, dc * 32, lane);
      s0 = wmma_bf16(qf[dc], k0f, s0);
      s1 = wmma_bf16(qf[dc], k1f, s1);
    }

    // ---- online softmax: per-row max over 32 cols (16-lane shfl reduction)
    f32x8 tmax;
#pragma unroll
    for (int e = 0; e < 8; ++e) tmax[e] = fmaxf(s0[e], s1[e]);
#pragma unroll
    for (int m = 1; m < 16; m <<= 1)
#pragma unroll
      for (int e = 0; e < 8; ++e)
        tmax[e] = fmaxf(tmax[e], __shfl_xor(tmax[e], m, 32));

    f32x8 mnew, alpha;
#pragma unroll
    for (int e = 0; e < 8; ++e) {
      mnew[e]  = fmaxf(mrun[e], tmax[e]);
      alpha[e] = __expf(mrun[e] - mnew[e]);
    }
#pragma unroll
    for (int f = 0; f < 4; ++f)
#pragma unroll
      for (int e = 0; e < 8; ++e) of[f][e] *= alpha[e];

    f32x8 p0, p1, rs;
#pragma unroll
    for (int e = 0; e < 8; ++e) {
      p0[e] = __expf(s0[e] - mnew[e]);
      p1[e] = __expf(s1[e] - mnew[e]);
      rs[e] = p0[e] + p1[e];
    }
#pragma unroll
    for (int m = 1; m < 16; m <<= 1)
#pragma unroll
      for (int e = 0; e < 8; ++e) rs[e] += __shfl_xor(rs[e], m, 32);
#pragma unroll
    for (int e = 0; e < 8; ++e) {
      lsum[e] = lsum[e] * alpha[e] + rs[e];
      mrun[e] = mnew[e];
    }

    // ---- C-layout P tile -> LDS (row-major bf16) -> A-fragment layout.
    // Producer and consumer are the SAME wave: dep handled via s_wait_dscnt.
    const int cb     = lane & 15;
    const int rowoff = (lane >> 4) * 8;
#pragma unroll
    for (int e = 0; e < 8; ++e) {
      lds_p[wv][rowoff + e][cb]      = f32_to_bf16(p0[e]);
      lds_p[wv][rowoff + e][16 + cb] = f32_to_bf16(p1[e]);
    }
    u16x16 pf = load_a_frag(&lds_p[wv][0][0], 32, 0, 0, lane);

    // ---- out += P(16x32) @ V(32x64); V transposed so B-frags are contiguous
#pragma unroll
    for (int nt = 0; nt < 4; ++nt) {
      u16x16 vf = load_b_frag(Vb, SLEN, nt * 16, key0, lane);
      of[nt] = wmma_bf16(pf, vf, of[nt]);
    }
  }

  // ---- normalize by softmax denominator and store bf16 attention output
  const int cb     = lane & 15;
  const int rowoff = (lane >> 4) * 8;
#pragma unroll
  for (int e = 0; e < 8; ++e) {
    const float inv = 1.0f / lsum[e];
    const size_t rowbase = (size_t)(b * SLEN + q0 + rowoff + e) * DDIM + h * DHEAD;
#pragma unroll
    for (int f = 0; f < 4; ++f)
      O[rowbase + f * 16 + cb] = f32_to_bf16(of[f][e] * inv);
  }
}

// ---------------------------------------------------------------------------
// Host-side launcher
// ---------------------------------------------------------------------------
extern "C" void kernel_launch(void* const* d_in, const int* in_sizes, int n_in,
                              void* d_out, int out_size, void* d_ws, size_t ws_size,
                              hipStream_t stream) {
  (void)in_sizes; (void)n_in; (void)out_size; (void)ws_size;

  const float* x   = (const float*)d_in[0];   // (2,2048,1024)
  const float* Wq  = (const float*)d_in[1];   // (1024,1024)
  const float* Wk  = (const float*)d_in[2];
  const float* Wv  = (const float*)d_in[3];
  const float* Wff = (const float*)d_in[4];
  const float* bff = (const float*)d_in[5];   // (1024,)
  float* out = (float*)d_out;                 // (2,2048,1024) fp32

  // Workspace layout (all bf16-as-u16): ~48 MiB total
  unsigned short* ws   = (unsigned short*)d_ws;
  const size_t NX = (size_t)MTOT * DDIM;      // 4096*1024
  const size_t NW = (size_t)DDIM * DDIM;      // 1024*1024
  unsigned short* xb   = ws;
  unsigned short* wqb  = xb   + NX;
  unsigned short* wkb  = wqb  + NW;
  unsigned short* wvb  = wkb  + NW;
  unsigned short* wffb = wvb  + NW;
  unsigned short* Qb   = wffb + NW;
  unsigned short* Kb   = Qb   + NX;
  unsigned short* Vtb  = Kb   + NX;           // [B][H][Dh][S]
  unsigned short* Ob   = Vtb  + NX;

  // 1) fp32 -> bf16
  cvt_bf16_kernel<<<2048, 256, 0, stream>>>(x,   xb,   (int)NX);
  cvt_bf16_kernel<<<1024, 256, 0, stream>>>(Wq,  wqb,  (int)NW);
  cvt_bf16_kernel<<<1024, 256, 0, stream>>>(Wk,  wkb,  (int)NW);
  cvt_bf16_kernel<<<1024, 256, 0, stream>>>(Wv,  wvb,  (int)NW);
  cvt_bf16_kernel<<<1024, 256, 0, stream>>>(Wff, wffb, (int)NW);

  // 2) projections: Q (scaled by 1/sqrt(D)=1/32), K, V (stored transposed)
  dim3 gg(MTOT / 128, DDIM / 128);
  gemm_bf16_kernel<<<gg, 256, 0, stream>>>(xb, wqb, nullptr, Qb,  nullptr,
                                           MTOT, DDIM, DDIM, 0, 0.03125f);
  gemm_bf16_kernel<<<gg, 256, 0, stream>>>(xb, wkb, nullptr, Kb,  nullptr,
                                           MTOT, DDIM, DDIM, 0, 1.0f);
  gemm_bf16_kernel<<<gg, 256, 0, stream>>>(xb, wvb, nullptr, Vtb, nullptr,
                                           MTOT, DDIM, DDIM, 1, 1.0f);

  // 3) flash attention
  flash_attn_kernel<<<dim3(SLEN / 128, HNUM, BATCH), 256, 0, stream>>>(Qb, Kb,
                                                                       Vtb, Ob);

  // 4) final feed-forward with bias -> fp32 output
  gemm_bf16_kernel<<<gg, 256, 0, stream>>>(Ob, wffb, bff, nullptr, out,
                                           MTOT, DDIM, DDIM, 2, 1.0f);
}
